// GroupMoE_56160992362640
// MI455X (gfx1250) — matrix operand order repaired and checked by reading, not compiled, not run on hardware
//
#include <hip/hip_runtime.h>
#include <stdint.h>

// ---------------- problem constants (match reference) ----------------
#define B_TOK 4096
#define D_IN  1024
#define H_DIM 2048
#define O_DIM 1024
#define NE    8
#define ROWPAD 256   // slack rows in slot-indexed buffers

typedef __attribute__((ext_vector_type(16))) __bf16 v16bf;
typedef __attribute__((ext_vector_type(8)))  float  v8f;

union Frag { unsigned int u[8]; v16bf v; };

__device__ __forceinline__ unsigned short f2bf(float f) {
  unsigned int u = __float_as_uint(f);
  unsigned int r = u + 0x7fffu + ((u >> 16) & 1u);   // round-to-nearest-even
  return (unsigned short)(r >> 16);
}

struct XPtrs { const float* p[NE]; };

// ---------------- small utility kernels ----------------
__global__ void init_counts(int* counts, int* fill) {
  if (threadIdx.x < NE) { counts[threadIdx.x] = 0; fill[threadIdx.x] = 0; }
}

__global__ void cvt_x(XPtrs xp, unsigned short* __restrict__ d) {
  const long n = (long)NE * B_TOK * D_IN;           // 8 * 2^22
  long i  = (long)blockIdx.x * blockDim.x + threadIdx.x;
  long st = (long)gridDim.x * blockDim.x;
  for (; i < n; i += st) {
    int  e = (int)(i >> 22);                        // B*D == 2^22
    long r = i & ((1L << 22) - 1);
    d[i] = f2bf(xp.p[e][r]);
  }
}

// Tiled transpose + fp32->bf16: wT[e][n][k] = bf16(w[e][k][n]).
// k becomes the contiguous dim so WMMA B fragments (K,K+1 pairs) and async
// 16B DMA chunks are both unit-stride.
__global__ void cvt_transpose(const float* __restrict__ w, unsigned short* __restrict__ wT,
                              int K, int N) {
  __shared__ float tile[32][33];
  int e  = blockIdx.z;
  int k0 = blockIdx.y * 32;
  int n0 = blockIdx.x * 32;
  const float* src = w + (size_t)e * K * N;
  unsigned short* dst = wT + (size_t)e * K * N;
  int tx = threadIdx.x;        // 0..31
  int ty = threadIdx.y;        // 0..7
  for (int r = ty; r < 32; r += 8)
    tile[r][tx] = src[(size_t)(k0 + r) * N + (n0 + tx)];
  __syncthreads();
  for (int r = ty; r < 32; r += 8)
    dst[(size_t)(n0 + r) * K + (k0 + tx)] = f2bf(tile[tx][r]);
}

// ---------------- gating: logits -> top-2 softmax -> counts ----------------
__global__ void gate_topk(XPtrs xp, const float* __restrict__ gw, const float* __restrict__ gb,
                          int* __restrict__ tok_e, float* __restrict__ tok_w,
                          int* __restrict__ counts) {
  int wave = threadIdx.x >> 5;
  int lane = threadIdx.x & 31;
  int b = blockIdx.x * (blockDim.x >> 5) + wave;
  if (b >= B_TOK) return;

  float acc[NE];
#pragma unroll
  for (int c = 0; c < NE; c++) acc[c] = 0.f;

  for (int idx = lane; idx < NE * D_IN; idx += 32) {
    int i  = idx >> 10;
    int dd = idx & (D_IN - 1);
    float xv = xp.p[i][(long)b * D_IN + dd];
    const float* gr = gw + (long)idx * NE;
#pragma unroll
    for (int c = 0; c < NE; c++) acc[c] += xv * gr[c];
  }
#pragma unroll
  for (int c = 0; c < NE; c++) {
#pragma unroll
    for (int off = 16; off >= 1; off >>= 1) acc[c] += __shfl_xor(acc[c], off, 32);
  }
  if (lane == 0) {
    float lg[NE];
#pragma unroll
    for (int c = 0; c < NE; c++) lg[c] = acc[c] + gb[c];
    int e0 = 0; float v0 = lg[0];
#pragma unroll
    for (int c = 1; c < NE; c++) if (lg[c] > v0) { v0 = lg[c]; e0 = c; }
    int e1 = (e0 == 0) ? 1 : 0; float v1 = lg[e1];
#pragma unroll
    for (int c = 0; c < NE; c++) if (c != e0 && lg[c] > v1) { v1 = lg[c]; e1 = c; }
    float w0 = 1.f, w1 = __expf(v1 - v0);             // softmax over {v0,v1}, v0 max
    float s = w0 + w1;
    tok_e[2 * b] = e0; tok_e[2 * b + 1] = e1;
    tok_w[2 * b] = w0 / s; tok_w[2 * b + 1] = w1 / s;
    atomicAdd(&counts[e0], 1);
    atomicAdd(&counts[e1], 1);
  }
}

__global__ void scan_counts(const int* __restrict__ counts, int* __restrict__ seg) {
  if (threadIdx.x == 0 && blockIdx.x == 0) {
    int s = 0;
    for (int e = 0; e < NE; e++) { seg[e] = s; s += counts[e]; }
    seg[NE] = s;
  }
}

__global__ void assign_slots(const int* __restrict__ tok_e, const int* __restrict__ seg,
                             int* __restrict__ fill, int* __restrict__ rowlist,
                             int* __restrict__ token_slot) {
  int b = blockIdx.x * blockDim.x + threadIdx.x;
  if (b >= B_TOK) return;
#pragma unroll
  for (int j = 0; j < 2; j++) {
    int e = tok_e[2 * b + j];
    int pos = atomicAdd(&fill[e], 1);
    int slot = seg[e] + pos;
    rowlist[slot] = b;
    token_slot[2 * b + j] = slot;
  }
}

// ---------------- routed bf16 WMMA GEMM: C = act(A @ WT^T + bias) ----------------
// 256 thr / 8 waves; block tile 128x128; wave tile 32x64 (2x4 WMMA frags).
// Triple-buffered LDS tiles filled by async global->LDS DMA (ASYNCcnt),
// one workgroup barrier per k-step.
__launch_bounds__(256)
__global__ void moe_gemm(const unsigned short* __restrict__ Abase, int lda, long aExpertStride,
                         const int* __restrict__ rowlist, int gather,
                         const unsigned short* __restrict__ WTbase, int K, int N,
                         const float* __restrict__ biasBase,
                         const int* __restrict__ seg, const int* __restrict__ counts,
                         unsigned short* __restrict__ CoutBf, float* __restrict__ CoutF,
                         int ldc, int relu) {
  __shared__ __align__(16) unsigned short lA[3][128 * 40];  // padded stride: bank-conflict free
  __shared__ __align__(16) unsigned short lB[3][128 * 40];  // WT tile [n][k], k-pairs contiguous
  constexpr unsigned BUFB = 128 * 40 * 2;                   // 10240 B per buffer

  const int e = blockIdx.z;
  const int cnt = counts[e];
  const int rowBase = blockIdx.y * 128;
  if (rowBase >= cnt) return;                 // routed: most tiles exit here
  const int segOff = seg[e];

  const int t = threadIdx.x;
  const int lane = t & 31;
  const int w = t >> 5;
  const int wr = w >> 1, wc = w & 1;          // 4x2 wave grid
  const int m = lane & 15, hi = lane >> 4;

  // ---- async staging setup: per wave, per tile: 2 A-ops + 2 B-ops, 512B each ----
  // op j covers 8 rows: row = w*16 + j*8 + (lane>>2); lane&3 picks the 16B chunk of a 64B row.
  const int chunk = lane & 3;
  const unsigned short* Aexp = Abase + aExpertStride * e;
  const unsigned short* WT   = WTbase + (long)e * K * N;
  unsigned aOff[2], aLds[2], bOff[2], bLds[2];
#pragma unroll
  for (int j = 0; j < 2; j++) {
    int r  = w * 16 + j * 8 + (lane >> 2);
    int lr = rowBase + r;
    int rg = segOff + ((lr < cnt) ? lr : (cnt - 1));      // clamp inside segment
    long rowelem;
    if (gather) rowelem = (long)(rowlist[rg] & (B_TOK - 1)) * lda;
    else        rowelem = (long)rg * lda;
    aOff[j] = (unsigned)((rowelem + chunk * 8) * 2);
    aLds[j] = (unsigned)(uintptr_t)&lA[0][r * 40 + chunk * 8];
    bOff[j] = (unsigned)(((long)(blockIdx.x * 128 + r) * K + chunk * 8) * 2);
    bLds[j] = (unsigned)(uintptr_t)&lB[0][r * 40 + chunk * 8];
  }
  const float* bias = biasBase + (long)e * N;

  const int T = K / 32;
  auto issue = [&](int ti, int buf) {
    unsigned kb = (unsigned)ti * 64u;          // 32 k-elems * 2B per tile step
    unsigned bo = (unsigned)buf * BUFB;
#pragma unroll
    for (int j = 0; j < 2; j++) {
      asm volatile("global_load_async_to_lds_b128 %0, %1, %2"
                   :: "v"(aLds[j] + bo), "v"(aOff[j] + kb), "s"(Aexp) : "memory");
      asm volatile("global_load_async_to_lds_b128 %0, %1, %2"
                   :: "v"(bLds[j] + bo), "v"(bOff[j] + kb), "s"(WT) : "memory");
    }
  };

  issue(0, 0);
  if (T > 1) issue(1, 1);

  const v8f vzero = {0.f, 0.f, 0.f, 0.f, 0.f, 0.f, 0.f, 0.f};
  v8f acc[2][4];
#pragma unroll
  for (int fr = 0; fr < 2; fr++)
#pragma unroll
    for (int fc = 0; fc < 4; fc++) acc[fr][fc] = vzero;

  for (int i = 0; i < T; i++) {
    // own chunks of tile i complete (in-order: <=4 leaves only tile i+1 in flight)
    if (i + 1 < T) asm volatile("s_wait_asynccnt 4" ::: "memory");
    else           asm volatile("s_wait_asynccnt 0" ::: "memory");
    __syncthreads();   // tile i fully in LDS; all waves done reading buf[(i-1)%3]
    if (i + 2 < T) issue(i + 2, (i + 2) % 3);   // overwrites buf[(i-1)%3]: safe after barrier

    const unsigned short* As = lA[i % 3];
    const unsigned short* Bs = lB[i % 3];

    Frag af[2], bfr[4];
#pragma unroll
    for (int fr = 0; fr < 2; fr++) {
      int r = wr * 32 + fr * 16 + m;
#pragma unroll
      for (int v = 0; v < 8; v++) {
        int kk = hi * 16 + v * 2;                             // contiguous k = 16*hi + s (matches B)
        af[fr].u[v] = *(const unsigned int*)&As[r * 40 + kk];
      }
    }
#pragma unroll
    for (int fc = 0; fc < 4; fc++) {
      int c = wc * 64 + fc * 16 + m;
#pragma unroll
      for (int v = 0; v < 8; v++) {
        int kk = hi * 16 + v * 2;                             // contiguous k = 16*hi + s (matches A)
        bfr[fc].u[v] = *(const unsigned int*)&Bs[c * 40 + kk];
      }
    }
#pragma unroll
    for (int fr = 0; fr < 2; fr++)
#pragma unroll
      for (int fc = 0; fc < 4; fc++)
        acc[fr][fc] = __builtin_amdgcn_wmma_f32_16x16x32_bf16(
            false, af[fr].v, false, bfr[fc].v, (short)0, acc[fr][fc], false, false);
  }

  // epilogue: bias (+ReLU), masked store. C layout: m = v + hi*8, n = lane&15.
#pragma unroll
  for (int fr = 0; fr < 2; fr++) {
#pragma unroll
    for (int fc = 0; fc < 4; fc++) {
      int col = blockIdx.x * 128 + wc * 64 + fc * 16 + m;
      float bv = bias[col];
#pragma unroll
      for (int v = 0; v < 8; v++) {
        int rl = rowBase + wr * 32 + fr * 16 + hi * 8 + v;
        if (rl < cnt) {
          float val = acc[fr][fc][v] + bv;
          if (relu) val = val > 0.f ? val : 0.f;
          long offo = (long)(segOff + rl) * ldc + col;
          if (CoutBf) CoutBf[offo] = f2bf(val);
          else        CoutF[offo]  = val;
        }
      }
    }
  }
}

// ---------------- deterministic weighted combine (fixed 2-slot sum) ----------------
__global__ void combine_out(const float* __restrict__ eout, const int* __restrict__ token_slot,
                            const float* __restrict__ tok_w, float* __restrict__ out) {
  int b = blockIdx.x;
  int o = blockIdx.y * blockDim.x + threadIdx.x;
  int s0 = token_slot[2 * b], s1 = token_slot[2 * b + 1];
  float w0 = tok_w[2 * b], w1 = tok_w[2 * b + 1];
  out[(long)b * O_DIM + o] = w0 * eout[(long)s0 * O_DIM + o] + w1 * eout[(long)s1 * O_DIM + o];
}

// ---------------- host-side orchestration ----------------
extern "C" void kernel_launch(void* const* d_in, const int* in_sizes, int n_in,
                              void* d_out, int out_size, void* d_ws, size_t ws_size,
                              hipStream_t stream) {
  (void)in_sizes; (void)n_in; (void)out_size; (void)ws_size;
  XPtrs xp;
  for (int i = 0; i < NE; i++) xp.p[i] = (const float*)d_in[i];
  const float* gate_w = (const float*)d_in[8];
  const float* gate_b = (const float*)d_in[9];
  const float* w_in   = (const float*)d_in[10];
  const float* b_in   = (const float*)d_in[11];
  const float* w_h    = (const float*)d_in[12];
  const float* b_h    = (const float*)d_in[13];
  const float* w_out  = (const float*)d_in[14];
  const float* b_out  = (const float*)d_in[15];
  float* out = (float*)d_out;

  // workspace carve-out (~305 MB total)
  char* ws = (char*)d_ws;
  size_t off = 0;
  auto alloc = [&](size_t bytes) -> char* {
    char* p = ws + off;
    off = (off + bytes + 255) & ~(size_t)255;
    return p;
  };
  int*   counts  = (int*)alloc(NE * 4);
  int*   fill    = (int*)alloc(NE * 4);
  int*   seg     = (int*)alloc((NE + 1) * 4);
  int*   tok_e   = (int*)alloc((size_t)B_TOK * 2 * 4);
  float* tokw    = (float*)alloc((size_t)B_TOK * 2 * 4);
  int*   tslot   = (int*)alloc((size_t)B_TOK * 2 * 4);
  int*   rowlist = (int*)alloc((size_t)(2 * B_TOK + ROWPAD) * 4);
  unsigned short* xbf  = (unsigned short*)alloc((size_t)NE * B_TOK * D_IN * 2);
  unsigned short* wibf = (unsigned short*)alloc((size_t)NE * D_IN * H_DIM * 2);   // transposed [N][K]
  unsigned short* whbf = (unsigned short*)alloc((size_t)NE * H_DIM * H_DIM * 2);  // transposed
  unsigned short* wobf = (unsigned short*)alloc((size_t)NE * H_DIM * O_DIM * 2);  // transposed
  unsigned short* h1   = (unsigned short*)alloc((size_t)(2 * B_TOK + ROWPAD) * H_DIM * 2);
  unsigned short* h2   = (unsigned short*)alloc((size_t)(2 * B_TOK + ROWPAD) * H_DIM * 2);
  float*          eout = (float*)alloc((size_t)(2 * B_TOK + ROWPAD) * O_DIM * 4);

  hipLaunchKernelGGL(init_counts, dim3(1), dim3(32), 0, stream, counts, fill);
  hipLaunchKernelGGL(cvt_x, dim3(4096), dim3(256), 0, stream, xp, xbf);
  hipLaunchKernelGGL(cvt_transpose, dim3(H_DIM / 32, D_IN / 32, NE),  dim3(32, 8), 0, stream, w_in,  wibf, D_IN, H_DIM);
  hipLaunchKernelGGL(cvt_transpose, dim3(H_DIM / 32, H_DIM / 32, NE), dim3(32, 8), 0, stream, w_h,   whbf, H_DIM, H_DIM);
  hipLaunchKernelGGL(cvt_transpose, dim3(O_DIM / 32, H_DIM / 32, NE), dim3(32, 8), 0, stream, w_out, wobf, H_DIM, O_DIM);
  hipLaunchKernelGGL(gate_topk, dim3(B_TOK / 8), dim3(256), 0, stream, xp, gate_w, gate_b, tok_e, tokw, counts);
  hipLaunchKernelGGL(scan_counts, dim3(1), dim3(1), 0, stream, counts, seg);
  hipLaunchKernelGGL(assign_slots, dim3(B_TOK / 256), dim3(256), 0, stream, tok_e, seg, fill, rowlist, tslot);

  // stage 1: h1 = relu(gather(x_e) @ w_in[e] + b_in[e])   [cnt_e, D] x [D, H]
  hipLaunchKernelGGL(moe_gemm, dim3(H_DIM / 128, B_TOK / 128, NE), dim3(256), 0, stream,
                     xbf, D_IN, (long)B_TOK * D_IN, rowlist, 1,
                     wibf, D_IN, H_DIM, b_in, seg, counts,
                     h1, (float*)nullptr, H_DIM, 1);
  // stage 2: h2 = relu(h1 @ w_h[e] + b_h[e])              [cnt_e, H] x [H, H]
  hipLaunchKernelGGL(moe_gemm, dim3(H_DIM / 128, B_TOK / 128, NE), dim3(256), 0, stream,
                     h1, H_DIM, 0L, (const int*)nullptr, 0,
                     whbf, H_DIM, H_DIM, b_h, seg, counts,
                     h2, (float*)nullptr, H_DIM, 1);
  // stage 3: eout = h2 @ w_out[e] + b_out[e]  (fp32 out)  [cnt_e, H] x [H, O]
  hipLaunchKernelGGL(moe_gemm, dim3(O_DIM / 128, B_TOK / 128, NE), dim3(256), 0, stream,
                     h2, H_DIM, 0L, (const int*)nullptr, 0,
                     wobf, H_DIM, O_DIM, b_out, seg, counts,
                     (unsigned short*)nullptr, eout, O_DIM, 0);

  hipLaunchKernelGGL(combine_out, dim3(B_TOK, O_DIM / 256), dim3(256), 0, stream, eout, tslot, tokw, out);
}
